// SiGNN_26465588478206
// MI455X (gfx1250) — matrix operand order repaired and checked by reading, not compiled, not run on hardware
//
#include <hip/hip_runtime.h>

// ---------- types for WMMA ----------
typedef __attribute__((ext_vector_type(16))) __bf16       v16bf;
typedef __attribute__((ext_vector_type(2)))  __bf16       v2bf;
typedef __attribute__((ext_vector_type(8)))  float        v8f;
typedef __attribute__((ext_vector_type(8)))  unsigned int v8u;

union FragU { v16bf h; v8u u; };

// ---------- bf16 helpers ----------
static __device__ __forceinline__ unsigned short f2bf(float x) {
    unsigned int u = __float_as_uint(x);
    u = (u + 0x7FFFu + ((u >> 16) & 1u)) >> 16;
    return (unsigned short)u;
}
static __device__ __forceinline__ float bf2f_lo(unsigned int w) {
    return __uint_as_float(w << 16);
}
static __device__ __forceinline__ float bf2f_hi(unsigned int w) {
    return __uint_as_float(w & 0xFFFF0000u);
}
// pack two f32 -> one dword of two bf16 (RNE)
static __device__ __forceinline__ unsigned int pkbf(float a, float b) {
#if defined(__has_builtin) && __has_builtin(__builtin_amdgcn_cvt_pk_bf16_f32)
    union { v2bf h; unsigned int u; } cv;
    cv.h = __builtin_amdgcn_cvt_pk_bf16_f32(a, b);
    return cv.u;
#else
    return (unsigned int)f2bf(a) | ((unsigned int)f2bf(b) << 16);
#endif
}

// ---------- fragment loaders ----------
// A-matrix 16x32 bf16 layout (ISA 7.12.2): lane L in [0,15] holds row M=L,
// lane L+16 same rows, other K half-groups:
//   VGPR p in 0..3: K = hi*8 + 2p ; VGPR p in 4..7: K = 16 + hi*8 + 2(p-4)
static __device__ __forceinline__ FragU load_frag_a(const unsigned int* w32,
                                                    int rowbase, int kc, int lane) {
    const int r   = rowbase + (lane & 15);
    const int hib = (lane >> 4) << 3;
    FragU f;
#pragma unroll
    for (int p = 0; p < 8; ++p) {
        int kk = ((p & 4) << 2) + hib + ((p & 3) << 1);
        f.u[p] = w32[r * 64 + ((kc + kk) >> 1)];        // row = 64 dwords
    }
    return f;
}
// B fragments pre-swizzled in d_ws: 1KB per fragment, lane-major 8 dwords/lane.
static __device__ __forceinline__ FragU load_frag_b(const unsigned int* ws,
                                                    int fid, int lane) {
    FragU f;
    f.u = *reinterpret_cast<const v8u*>(ws + fid * 256 + lane * 8);
    return f;
}

#define WMMA_BF16(A, B, C) \
    __builtin_amdgcn_wmma_f32_16x16x32_bf16(false, (A).h, false, (B).h, (short)0, (C), false, false)

#define NFAM   100000
#define FAMPB  16
#define NBLK   (NFAM / FAMPB)

// ======================================================================
// Prep: swizzle the 8 weight matrices (f32 row-major [H,128]) into the
// per-lane B-fragment layout. Layer1: fid = m*32 + nt*4 + kc (m: Wl1,Wr1,
// Wlt1,Wrt1; 8 ntiles). Layer2: fid = 128 + m*16 + nt*4 + kc (4 ntiles).
// B[k][n] = W[n][k]; lane = n (+16 for hi); VGPR p: K = kc*32+(p>=4?16:0)+hi*8+2(p&3).
// ======================================================================
__global__ void __launch_bounds__(256) signn_prep(
    const float* __restrict__ Wl1,  const float* __restrict__ Wr1,
    const float* __restrict__ Wlt1, const float* __restrict__ Wrt1,
    const float* __restrict__ Wl2,  const float* __restrict__ Wr2,
    const float* __restrict__ Wlt2, const float* __restrict__ Wrt2,
    unsigned int* __restrict__ ws)
{
    const int fid  = blockIdx.x * 8 + (threadIdx.x >> 5);   // 0..191
    const int lane = threadIdx.x & 31;
    const float* mats[8] = {Wl1, Wr1, Wlt1, Wrt1, Wl2, Wr2, Wlt2, Wrt2};

    int m, ntile, kci;
    if (fid < 128) { m = fid >> 5;              ntile = (fid >> 2) & 7; kci = fid & 3; }
    else           { int g = fid - 128; m = 4 + (g >> 4); ntile = (g >> 2) & 3; kci = g & 3; }

    const float* W = mats[m];
    const int n  = ntile * 16 + (lane & 15);
    const int hi = lane >> 4;
    unsigned int* dst = ws + fid * 256 + lane * 8;
#pragma unroll
    for (int p = 0; p < 8; ++p) {
        int k = kci * 32 + ((p & 4) << 2) + hi * 8 + 2 * (p & 3);
        dst[p] = pkbf(W[n * 128 + k], W[n * 128 + k + 1]);
    }
}

// ======================================================================
// Main fused kernel: one block = 16 families (16 h0 + 80 h1 + 160 h2 rows).
// LDS buffers held as dwords (2 bf16 per dword, 64 dwords per 128-col row).
// ======================================================================
__global__ void __launch_bounds__(256) signn_main(
    const float* __restrict__ h0, const float* __restrict__ h1,
    const float* __restrict__ h2, const unsigned int* __restrict__ ws,
    float* __restrict__ out)
{
    __shared__ unsigned int Xs32[96 * 64];   // x_self (bf16x2) -> later neigh2 rows 0..15
    __shared__ unsigned int Ng32[96 * 64];   // neigh  (bf16x2) -> later out_s

    const int tid  = threadIdx.x;
    const int lane = tid & 31;
    const int wave = tid >> 5;
    const int fam0 = blockIdx.x * FAMPB;

    // ---- stage 1: stream inputs (each byte read once, coalesced b64/lane) ----
    {   // h0 -> Xs rows 0..15 (1024 dwords)
        const size_t b = (size_t)fam0 * 128;
#pragma unroll
        for (int j = 0; j < 4; ++j) {
            int w = tid + 256 * j;
            Xs32[w] = pkbf(h0[b + 2 * w], h0[b + 2 * w + 1]);
        }
    }
    {   // h1 -> Xs rows 16..95 (5120 dwords)
        const size_t b = (size_t)fam0 * 5 * 128;
        for (int j = 0; j < 20; ++j) {
            int w = tid + 256 * j;
            Xs32[1024 + w] = pkbf(h1[b + 2 * w], h1[b + 2 * w + 1]);
        }
    }
    {   // h2 pair means -> Ng rows 16..95 (5120 dwords)
        const size_t b = (size_t)fam0 * 10 * 128;
        for (int j = 0; j < 20; ++j) {
            int w  = tid + 256 * j;
            int r  = w >> 6, c2 = (w & 63) * 2;
            size_t g = b + (size_t)(2 * r) * 128 + c2;
            Ng32[1024 + w] = pkbf(0.5f * (h2[g]     + h2[g + 128]),
                                  0.5f * (h2[g + 1] + h2[g + 129]));
        }
    }
    __syncthreads();
    // neigh1 = mean of 5 h1 rows (from LDS) -> Ng rows 0..15
#pragma unroll
    for (int j = 0; j < 4; ++j) {
        int w = tid + 256 * j;
        int i = w >> 6, cp = w & 63;
        float sl = 0.f, sh = 0.f;
#pragma unroll
        for (int q = 0; q < 5; ++q) {
            unsigned int v = Xs32[1024 + (i * 5 + q) * 64 + cp];
            sl += bf2f_lo(v); sh += bf2f_hi(v);
        }
        Ng32[w] = pkbf(0.2f * sl, 0.2f * sh);
    }
    __syncthreads();

    // ---- stage 2: layer 1 — wave w owns column tile ntile=w; kci-outer loop
    //      register-caches the 4 B fragments across all 6 row tiles. ----
    v8f accT[6], accX[6];
#pragma unroll
    for (int rt = 0; rt < 6; ++rt) { accT[rt] = (v8f){}; accX[rt] = (v8f){}; }

#pragma unroll
    for (int kci = 0; kci < 4; ++kci) {
        FragU bl = load_frag_b(ws,  0 + wave * 4 + kci, lane);  // Wl1
        FragU br = load_frag_b(ws, 32 + wave * 4 + kci, lane);  // Wr1
        FragU bt = load_frag_b(ws, 64 + wave * 4 + kci, lane);  // Wlt1
        FragU bq = load_frag_b(ws, 96 + wave * 4 + kci, lane);  // Wrt1
#pragma unroll
        for (int rt = 0; rt < 6; ++rt) {
            FragU as = load_frag_a(Xs32, rt * 16, kci * 32, lane);
            FragU an = load_frag_a(Ng32, rt * 16, kci * 32, lane);
            accX[rt] = WMMA_BF16(as, bl, accX[rt]);
            accX[rt] = WMMA_BF16(an, br, accX[rt]);
            accT[rt] = WMMA_BF16(as, bt, accT[rt]);
            accT[rt] = WMMA_BF16(an, bq, accT[rt]);
        }
    }
    // activation in-register: accX[rt][r] := sigmoid(x) * (t >= 1)
#pragma unroll
    for (int rt = 0; rt < 6; ++rt)
#pragma unroll
        for (int r = 0; r < 8; ++r) {
            float sx = 1.0f / (1.0f + __expf(-accX[rt][r]));
            accX[rt][r] = (accT[rt][r] >= 1.0f) ? sx : 0.0f;
        }
    __syncthreads();   // all reads of Ng done -> safe to overwrite with out_s
    {
        unsigned short* Ngus = (unsigned short*)Ng32;
        const int n = wave * 16 + (lane & 15);
#pragma unroll
        for (int rt = 0; rt < 6; ++rt) {
            const int mb = rt * 16 + ((lane >> 4) << 3);
#pragma unroll
            for (int r = 0; r < 8; ++r)
                Ngus[(mb + r) * 128 + n] = f2bf(accX[rt][r]);
        }
    }
    __syncthreads();
    // neigh2 = mean of 5 h1p rows -> Xs rows 0..15
#pragma unroll
    for (int j = 0; j < 4; ++j) {
        int w = tid + 256 * j;
        int i = w >> 6, cp = w & 63;
        float sl = 0.f, sh = 0.f;
#pragma unroll
        for (int q = 0; q < 5; ++q) {
            unsigned int v = Ng32[1024 + (i * 5 + q) * 64 + cp];
            sl += bf2f_lo(v); sh += bf2f_hi(v);
        }
        Xs32[w] = pkbf(0.2f * sl, 0.2f * sh);
    }
    __syncthreads();

    // ---- stage 3: layer 2 (16x64 output, 4 coltiles on waves 0..3) ----
    if (wave < 4) {
        const int ntile = wave;
        v8f aT = {}; v8f aX = {};
#pragma unroll
        for (int kci = 0; kci < 4; ++kci) {
            FragU a0 = load_frag_a(Ng32, 0, kci * 32, lane);  // h0p
            FragU a1 = load_frag_a(Xs32, 0, kci * 32, lane);  // neigh2
            FragU bl = load_frag_b(ws, 128 +  0 + ntile * 4 + kci, lane);  // Wl2
            FragU br = load_frag_b(ws, 128 + 16 + ntile * 4 + kci, lane);  // Wr2
            FragU bt = load_frag_b(ws, 128 + 32 + ntile * 4 + kci, lane);  // Wlt2
            FragU bq = load_frag_b(ws, 128 + 48 + ntile * 4 + kci, lane);  // Wrt2
            aX = WMMA_BF16(a0, bl, aX);
            aX = WMMA_BF16(a1, br, aX);
            aT = WMMA_BF16(a0, bt, aT);
            aT = WMMA_BF16(a1, bq, aT);
        }
        const int mb = (lane >> 4) << 3;
        const int n  = lane & 15;
#pragma unroll
        for (int r = 0; r < 8; ++r) {
            float sx = 1.0f / (1.0f + __expf(-aX[r]));
            float v  = (aT[r] >= 1.0f) ? sx : 0.0f;
            out[(size_t)(fam0 + mb + r) * 64 + ntile * 16 + n] = v;
        }
    }
}

// ======================================================================
extern "C" void kernel_launch(void* const* d_in, const int* in_sizes, int n_in,
                              void* d_out, int out_size, void* d_ws, size_t ws_size,
                              hipStream_t stream)
{
    const float* h0   = (const float*)d_in[0];
    const float* h1   = (const float*)d_in[1];
    const float* h2   = (const float*)d_in[2];
    const float* Wl1  = (const float*)d_in[3];
    const float* Wr1  = (const float*)d_in[4];
    const float* Wlt1 = (const float*)d_in[5];
    const float* Wrt1 = (const float*)d_in[6];
    const float* Wl2  = (const float*)d_in[7];
    const float* Wr2  = (const float*)d_in[8];
    const float* Wlt2 = (const float*)d_in[9];
    const float* Wrt2 = (const float*)d_in[10];
    unsigned int* ws  = (unsigned int*)d_ws;   // needs 192 KB
    float* out        = (float*)d_out;

    signn_prep<<<24, 256, 0, stream>>>(Wl1, Wr1, Wlt1, Wrt1, Wl2, Wr2, Wlt2, Wrt2, ws);
    signn_main<<<NBLK, 256, 0, stream>>>(h0, h1, h2, ws, out);
}
